// Transliteration_EncoderDecoder_BeamSearch_90099823935718
// MI455X (gfx1250) — matrix operand order repaired and checked by reading, not compiled, not run on hardware
//
#include <hip/hip_runtime.h>
#include <hip/hip_bf16.h>
#include <math.h>

// ---------------------------------------------------------------------------
// Seq2seq RNN (encoder 3200 steps, decoder 30 steps) for MI455X / gfx1250.
// Strategy: bf16 WMMA matvecs (v_wmma_f32_16x16x32_bf16) with weights
// pre-packed into the CDNA5 A-fragment lane layout; persistent kernels with
// an atomic grid barrier carry the sequential recurrence; decoder does
// softmax/top-k/threefry-categorical on-device.
// ---------------------------------------------------------------------------

typedef __attribute__((ext_vector_type(16))) __bf16 v16bf;
typedef __attribute__((ext_vector_type(8)))  float  v8f;

#define HID   2048   // HIDDEN
#define EMBD  512    // EMB
#define OUTD  512    // OUTPUT
#define KCH   (HID / 32)   // 64 K-chunks of 32 for K=2048
#define GRID_P 64          // persistent grid: 64 blocks
#define BLK_P  64          // 2 waves per block -> 128 waves == 128 row tiles

// ---------------------------------------------------------------------------
// Threefry2x32 (JAX PRNG), 16 rounds.
// ---------------------------------------------------------------------------
__device__ __host__ __forceinline__ unsigned rotl32(unsigned x, int r) {
  return (x << r) | (x >> (32 - r));
}

__device__ __forceinline__ void threefry2x32(unsigned k0, unsigned k1,
                                             unsigned x0, unsigned x1,
                                             unsigned* o0, unsigned* o1) {
  const unsigned ks2 = 0x1BD11BDAu ^ k0 ^ k1;
  const int RA[4] = {13, 15, 26, 6};
  const int RB[4] = {17, 29, 16, 24};
  x0 += k0; x1 += k1;
#pragma unroll
  for (int r = 0; r < 4; ++r) { x0 += x1; x1 = rotl32(x1, RA[r]); x1 ^= x0; }
  x0 += k1; x1 += ks2 + 1u;
#pragma unroll
  for (int r = 0; r < 4; ++r) { x0 += x1; x1 = rotl32(x1, RB[r]); x1 ^= x0; }
  x0 += ks2; x1 += k0 + 2u;
#pragma unroll
  for (int r = 0; r < 4; ++r) { x0 += x1; x1 = rotl32(x1, RA[r]); x1 ^= x0; }
  x0 += k0; x1 += k1 + 3u;
#pragma unroll
  for (int r = 0; r < 4; ++r) { x0 += x1; x1 = rotl32(x1, RB[r]); x1 ^= x0; }
  x0 += k1; x1 += ks2 + 4u;
  *o0 = x0; *o1 = x1;
}

__device__ __forceinline__ float gumbel_from_bits(unsigned bits) {
  // JAX: u = bitcast(bits>>9 | 0x3f800000) - 1; uniform(minval=tiny); -log(-log(u))
  const float tiny = 1.17549435e-38f;
  union { unsigned u; float f; } cvt;
  cvt.u = (bits >> 9) | 0x3f800000u;
  float u01 = cvt.f - 1.0f;
  float u = u01 * (1.0f - tiny) + tiny;
  u = fmaxf(u, tiny);
  return -logf(-logf(u));
}

// ---------------------------------------------------------------------------
// Grid-wide barrier: monotonic arrival counter (re-zeroed by prep each call).
// ---------------------------------------------------------------------------
__device__ __forceinline__ void grid_barrier(unsigned* cnt, unsigned target) {
  __syncthreads();
  if (threadIdx.x == 0) {
    __threadfence();
    __hip_atomic_fetch_add(cnt, 1u, __ATOMIC_RELEASE, __HIP_MEMORY_SCOPE_AGENT);
    while (__hip_atomic_load(cnt, __ATOMIC_ACQUIRE, __HIP_MEMORY_SCOPE_AGENT) < target) {
      __builtin_amdgcn_s_sleep(2);
    }
  }
  __syncthreads();
  __threadfence();
}

// ---------------------------------------------------------------------------
// One wave computes y[16] = W_tile(16 x 2048) . h(2048) via 64 bf16 WMMAs.
// Wp is pre-packed: flat index ((tile*KCH + chunk)*32 + lane)*16 + e, where
// element e holds K = 16*(lane/16) + e of the chunk (contiguous order). B operand = h
// broadcast to all 16 columns (lane loads 16 contiguous bf16 of h).
// Result columns are identical; caller extracts rows from lanes 0 / 16.
// ---------------------------------------------------------------------------
__device__ __forceinline__ v8f wave_matvec_tile(const __bf16* __restrict__ Wp,
                                                int tile,
                                                const __bf16* __restrict__ hin,
                                                int lane) {
  const v16bf* Ap = ((const v16bf*)Wp) + (size_t)tile * (KCH * 32);
  const __bf16* hb = hin + ((lane < 16) ? 0 : 16);
  v8f acc0 = {0.f, 0.f, 0.f, 0.f, 0.f, 0.f, 0.f, 0.f};
  v8f acc1 = {0.f, 0.f, 0.f, 0.f, 0.f, 0.f, 0.f, 0.f};
#pragma unroll 4
  for (int c = 0; c < KCH; c += 2) {
    if (c + 4 < KCH) __builtin_prefetch(&Ap[(c + 4) * 32 + lane], 0, 1);
    v16bf a0 = Ap[c * 32 + lane];
    v16bf b0 = *(const v16bf*)(hb + c * 32);
    acc0 = __builtin_amdgcn_wmma_f32_16x16x32_bf16(false, a0, false, b0,
                                                   (short)0, acc0, false, false);
    v16bf a1 = Ap[(c + 1) * 32 + lane];
    v16bf b1 = *(const v16bf*)(hb + (c + 1) * 32);
    acc1 = __builtin_amdgcn_wmma_f32_16x16x32_bf16(false, a1, false, b1,
                                                   (short)0, acc1, false, false);
  }
  v8f y;
#pragma unroll
  for (int i = 0; i < 8; ++i) y[i] = acc0[i] + acc1[i];
  return y;
}

// ---------------------------------------------------------------------------
// Pack fp32 row-major [rows x cols] into bf16 WMMA A-fragment layout.
// ---------------------------------------------------------------------------
__global__ void pack_bf16_wmma(const float* __restrict__ W, __bf16* __restrict__ P,
                               int rows, int cols) {
  int o = blockIdx.x * blockDim.x + threadIdx.x;
  int total = rows * cols;
  if (o >= total) return;
  int e    = o & 15;
  int lane = (o >> 4) & 31;
  int rest = o >> 9;                 // tile*chunkCnt + chunk
  int chunkCnt = cols >> 5;
  int chunk = rest % chunkCnt;
  int tile  = rest / chunkCnt;
  int row = tile * 16 + (lane & 15);
  // A-fragment K order: lanes 0-15 hold K {0..15}, lanes 16-31 {16..31}
  int k = ((lane >> 4) << 4) + e;
  int col = (chunk << 5) + k;
  P[o] = (__bf16)W[(size_t)row * cols + col];
}

// ---------------------------------------------------------------------------
// Prep: u[t] = enc_Wih . emb[t] + enc_bih + enc_bhh (t = 0,1),
//       dbias = dec_bih + dec_bhh, zero h ping-pong, reset control words,
//       threefry split(key(42), T) -> decoder step keys.
// ---------------------------------------------------------------------------
__global__ void prep_misc_kernel(const float* __restrict__ emb,
                                 const float* __restrict__ enc_Wih,
                                 const float* __restrict__ enc_bih,
                                 const float* __restrict__ enc_bhh,
                                 const float* __restrict__ dec_bih,
                                 const float* __restrict__ dec_bhh,
                                 float* __restrict__ u,
                                 float* __restrict__ dbias,
                                 __bf16* __restrict__ hbuf,
                                 unsigned* __restrict__ ctrl,
                                 unsigned* __restrict__ keys,
                                 int T) {
  int gid = blockIdx.x * blockDim.x + threadIdx.x;
  if (gid < 2 * HID) {
    int t = gid >> 11;               // 0 or 1
    int i = gid & (HID - 1);
    const float* wr = enc_Wih + (size_t)i * EMBD;
    const float* er = emb + (size_t)t * EMBD;
    float s = 0.f;
    for (int j = 0; j < EMBD; ++j) s += wr[j] * er[j];
    u[t * HID + i] = s + enc_bih[i] + enc_bhh[i];
  } else if (gid < 3 * HID) {
    int i = gid - 2 * HID;
    dbias[i] = dec_bih[i] + dec_bhh[i];
    hbuf[i] = (__bf16)0.0f;
    hbuf[HID + i] = (__bf16)0.0f;
  } else if (gid == 3 * HID) {
    ctrl[0] = 0u;                    // encoder barrier counter
    ctrl[1] = 0u;                    // decoder barrier counter
    ((int*)ctrl)[2] = -1;            // decoder feedback index (x0 = zeros)
    // jax.random.split(key(42), T): bits = threefry(counts 0..2T-1)
    unsigned bits[128];
    for (int i = 0; i < T; ++i) {
      unsigned a, b;
      threefry2x32(0u, 42u, (unsigned)i, (unsigned)(T + i), &a, &b);
      bits[i] = a; bits[T + i] = b;
    }
    for (int i = 0; i < 2 * T; ++i) keys[i] = bits[i];
  }
}

// ---------------------------------------------------------------------------
// Encoder: S sequential steps, whole grid does one 2048x2048 matvec per step.
// ---------------------------------------------------------------------------
__global__ void __launch_bounds__(BLK_P, 1)
enc_rnn_kernel(const int* __restrict__ tok, int S,
               const __bf16* __restrict__ Wp, const float* __restrict__ u,
               __bf16* __restrict__ hbuf, unsigned* __restrict__ cnt) {
  const int lane = threadIdx.x & 31;
  const int wid  = blockIdx.x * (BLK_P / 32) + (threadIdx.x >> 5);  // 0..127
  for (int s = 0; s < S; ++s) {
    const __bf16* hin  = hbuf + (s & 1) * HID;
    __bf16*       hout = hbuf + ((s + 1) & 1) * HID;
    v8f y = wave_matvec_tile(Wp, wid, hin, lane);
    if (lane == 0 || lane == 16) {
      const int rbase = wid * 16 + (lane >> 4) * 8;
      const int t = tok[s] & 1;
      const float* ub = u + t * HID;
#pragma unroll
      for (int r = 0; r < 8; ++r)
        hout[rbase + r] = (__bf16)tanhf(y[r] + ub[rbase + r]);
    }
    grid_barrier(cnt, (unsigned)(s + 1) * gridDim.x);
  }
}

// ---------------------------------------------------------------------------
// Decoder: T steps of  h = tanh(dec_Wih[:,idx] + dbias + dec_Whh.h),
// logits = h2o_W.h + h2o_b, log_softmax -> d_out, top-5 + threefry
// categorical -> next idx. Three grid-synced phases per step.
// ---------------------------------------------------------------------------
__global__ void __launch_bounds__(BLK_P, 1)
dec_rnn_kernel(int T, int parity,
               const __bf16* __restrict__ Wdp, const __bf16* __restrict__ Wop,
               const float* __restrict__ dWih, const float* __restrict__ dbias,
               const float* __restrict__ h2o_b, __bf16* __restrict__ hbuf,
               float* __restrict__ logits, float* __restrict__ out,
               const unsigned* __restrict__ keys, int* __restrict__ idxp,
               unsigned* __restrict__ cnt) {
  const int lane = threadIdx.x & 31;
  const int wid  = blockIdx.x * (BLK_P / 32) + (threadIdx.x >> 5);
  unsigned sync = 0;
  __shared__ float red[BLK_P];

  for (int t = 0; t < T; ++t) {
    const __bf16* hin  = hbuf + ((parity + t) & 1) * HID;
    __bf16*       hout = hbuf + ((parity + t + 1) & 1) * HID;

    // ---- phase 1: hidden recurrence (128 waves, one 16-row tile each) ----
    {
      v8f y = wave_matvec_tile(Wdp, wid, hin, lane);
      if (lane == 0 || lane == 16) {
        const int rbase = wid * 16 + (lane >> 4) * 8;
        const int xi = __hip_atomic_load(idxp, __ATOMIC_RELAXED,
                                         __HIP_MEMORY_SCOPE_AGENT);
#pragma unroll
        for (int r = 0; r < 8; ++r) {
          float v = y[r] + dbias[rbase + r];
          if (xi >= 0) v += dWih[(size_t)(rbase + r) * OUTD + xi];
          hout[rbase + r] = (__bf16)tanhf(v);
        }
      }
    }
    grid_barrier(cnt, ++sync * gridDim.x);

    // ---- phase 2: logits = h2o_W . h + b (waves 0..31 -> 512 rows) ----
    if (wid < OUTD / 16) {
      v8f y = wave_matvec_tile(Wop, wid, hout, lane);
      if (lane == 0 || lane == 16) {
        const int rbase = wid * 16 + (lane >> 4) * 8;
#pragma unroll
        for (int r = 0; r < 8; ++r)
          logits[rbase + r] = y[r] + h2o_b[rbase + r];
      }
    }
    grid_barrier(cnt, ++sync * gridDim.x);

    // ---- phase 3: log_softmax + top-5 + categorical (block 0) ----
    if (blockIdx.x == 0) {
      float m = -INFINITY;
      for (int i = threadIdx.x; i < OUTD; i += BLK_P) m = fmaxf(m, logits[i]);
      red[threadIdx.x] = m; __syncthreads();
      for (int st = BLK_P / 2; st > 0; st >>= 1) {
        if (threadIdx.x < st)
          red[threadIdx.x] = fmaxf(red[threadIdx.x], red[threadIdx.x + st]);
        __syncthreads();
      }
      const float mx = red[0]; __syncthreads();
      float ssum = 0.f;
      for (int i = threadIdx.x; i < OUTD; i += BLK_P) ssum += expf(logits[i] - mx);
      red[threadIdx.x] = ssum; __syncthreads();
      for (int st = BLK_P / 2; st > 0; st >>= 1) {
        if (threadIdx.x < st) red[threadIdx.x] += red[threadIdx.x + st];
        __syncthreads();
      }
      const float lz = mx + logf(red[0]);
      for (int i = threadIdx.x; i < OUTD; i += BLK_P)
        out[(size_t)t * OUTD + i] = logits[i] - lz;
      __syncthreads();

      if (threadIdx.x == 0) {
        // top-5 (stable descending, first index on ties == jax.lax.top_k)
        float topv[5]; int topi[5];
        for (int b = 0; b < 5; ++b) {
          float best = -INFINITY; int bi = 0;
          for (int i = 0; i < OUTD; ++i) {
            bool used = false;
            for (int p = 0; p < b; ++p) used |= (topi[p] == i);
            if (used) continue;
            float v = logits[i] - lz;
            if (v > best) { best = v; bi = i; }
          }
          topv[b] = best; topi[b] = bi;
        }
        float psum = 0.f;
        for (int b = 0; b < 5; ++b) {
          if (!__builtin_isfinite(topv[b])) topv[b] = 0.f;
          psum += topv[b];
        }
        // categorical(keys[t], log(probs)): gumbel bits for counts [0..4]
        unsigned k0 = keys[2 * t], k1 = keys[2 * t + 1];
        unsigned r00, r10, r01, r11, r02, r12;
        threefry2x32(k0, k1, 0u, 3u, &r00, &r10);
        threefry2x32(k0, k1, 1u, 4u, &r01, &r11);
        threefry2x32(k0, k1, 2u, 0u, &r02, &r12);
        unsigned bits5[5] = {r00, r01, r02, r10, r11};
        float bestg = -INFINITY; int sel = 0;
        for (int b = 0; b < 5; ++b) {
          float g = logf(topv[b] / psum) + gumbel_from_bits(bits5[b]);
          if (g > bestg) { bestg = g; sel = b; }
        }
        __hip_atomic_store(idxp, topi[sel], __ATOMIC_RELAXED,
                           __HIP_MEMORY_SCOPE_AGENT);
      }
    }
    grid_barrier(cnt, ++sync * gridDim.x);
  }
}

// ---------------------------------------------------------------------------
// Host launcher. Workspace (~19 MB): packed bf16 weights + small state.
// ---------------------------------------------------------------------------
extern "C" void kernel_launch(void* const* d_in, const int* in_sizes, int n_in,
                              void* d_out, int out_size, void* d_ws, size_t ws_size,
                              hipStream_t stream) {
  const int*   tokens  = (const int*)d_in[0];
  // d_in[1] = max_output_chars scalar (value 30); T derived from out_size.
  const float* emb     = (const float*)d_in[2];
  const float* enc_Wih = (const float*)d_in[3];
  const float* enc_Whh = (const float*)d_in[4];
  const float* enc_bih = (const float*)d_in[5];
  const float* enc_bhh = (const float*)d_in[6];
  const float* dec_Wih = (const float*)d_in[7];
  const float* dec_Whh = (const float*)d_in[8];
  const float* dec_bih = (const float*)d_in[9];
  const float* dec_bhh = (const float*)d_in[10];
  const float* h2o_W   = (const float*)d_in[11];
  const float* h2o_b   = (const float*)d_in[12];

  const int S = in_sizes[0];         // 3200 encoder steps (L * INPUT_SIZE)
  const int T = out_size / OUTD;     // 30 decoder steps

  char* ws = (char*)d_ws;
  size_t off = 0;
  auto take = [&](size_t bytes) -> void* {
    void* p = ws + off;
    off = (off + bytes + 255) & ~(size_t)255;
    return p;
  };
  __bf16*   Wenc_p = (__bf16*)take((size_t)HID * HID * 2);
  __bf16*   Wdec_p = (__bf16*)take((size_t)HID * HID * 2);
  __bf16*   Wh2o_p = (__bf16*)take((size_t)OUTD * HID * 2);
  float*    u      = (float*)take(2 * HID * sizeof(float));
  float*    dbias  = (float*)take(HID * sizeof(float));
  __bf16*   hbuf   = (__bf16*)take(2 * HID * 2);
  float*    logits = (float*)take(OUTD * sizeof(float));
  unsigned* keys   = (unsigned*)take(512);
  unsigned* ctrl   = (unsigned*)take(256);
  (void)ws_size; (void)n_in;

  pack_bf16_wmma<<<(HID * HID + 255) / 256, 256, 0, stream>>>(enc_Whh, Wenc_p, HID, HID);
  pack_bf16_wmma<<<(HID * HID + 255) / 256, 256, 0, stream>>>(dec_Whh, Wdec_p, HID, HID);
  pack_bf16_wmma<<<(OUTD * HID + 255) / 256, 256, 0, stream>>>(h2o_W, Wh2o_p, OUTD, HID);
  prep_misc_kernel<<<(3 * HID) / 256 + 2, 256, 0, stream>>>(
      emb, enc_Wih, enc_bih, enc_bhh, dec_bih, dec_bhh,
      u, dbias, hbuf, ctrl, keys, T);

  enc_rnn_kernel<<<GRID_P, BLK_P, 0, stream>>>(tokens, S, Wenc_p, u, hbuf, ctrl + 0);
  dec_rnn_kernel<<<GRID_P, BLK_P, 0, stream>>>(T, S & 1, Wdec_p, Wh2o_p, dec_Wih,
                                               dbias, h2o_b, hbuf, logits,
                                               (float*)d_out, keys,
                                               (int*)(ctrl + 2), ctrl + 1);
}
